// GPTModel_66159676228024
// MI455X (gfx1250) — matrix operand order repaired and checked by reading, not compiled, not run on hardware
//
#include <hip/hip_runtime.h>
#include <hip/hip_bf16.h>
#include <math.h>

// ---------------- model dims (GPT-2 124M, B=2, T=1024) ----------------
#define BATCH 2
#define SEQ   1024
#define DMODEL 768
#define NHEAD 12
#define HDIM  64
#define NLAYER 12
#define FFDIM 3072
#define VOCAB 50257
#define ROWS (BATCH * SEQ)          // 2048
#define LN_EPS 1e-5f

typedef __attribute__((ext_vector_type(16))) __bf16 bf16x16;
typedef __attribute__((ext_vector_type(8)))  __bf16 bf16x8;
typedef __attribute__((ext_vector_type(8)))  float  v8f;

// round-to-nearest-even f32 -> bf16 (bit-level, no reliance on cast lowering)
__device__ __forceinline__ __bf16 f2bf(float f) {
    unsigned u; __builtin_memcpy(&u, &f, 4);
    unsigned r = u + 0x7FFFu + ((u >> 16) & 1u);
    unsigned short hs = (unsigned short)(r >> 16);
    __bf16 b; __builtin_memcpy(&b, &hs, 2);
    return b;
}

// ---------------- embedding: x = wte[idx] + wpe[t] ----------------
__global__ __launch_bounds__(256)
void embed_kernel(const int* __restrict__ idx, const float* __restrict__ wte,
                  const float* __restrict__ wpe, float* __restrict__ x) {
    int row = blockIdx.x;                 // 0..ROWS-1
    int t = row % SEQ;
    int tok = idx[row];
    size_t ob = (size_t)row * DMODEL;
    size_t wb = (size_t)tok * DMODEL;
    size_t pb = (size_t)t * DMODEL;
    for (int i = threadIdx.x; i < DMODEL; i += 256)
        x[ob + i] = wte[wb + i] + wpe[pb + i];
}

// ---------------- layernorm (one block per row) ----------------
__global__ __launch_bounds__(256)
void layernorm_kernel(const float* __restrict__ x, const float* __restrict__ g,
                      const float* __restrict__ b, float* __restrict__ y) {
    __shared__ float r1[256], r2[256];
    int row = blockIdx.x;
    size_t base = (size_t)row * DMODEL;
    int tid = threadIdx.x;
    float s = 0.f, s2 = 0.f;
    for (int i = tid; i < DMODEL; i += 256) {
        float v = x[base + i];
        s += v; s2 += v * v;
    }
    r1[tid] = s; r2[tid] = s2;
    __syncthreads();
    for (int off = 128; off > 0; off >>= 1) {
        if (tid < off) { r1[tid] += r1[tid + off]; r2[tid] += r2[tid + off]; }
        __syncthreads();
    }
    float mu = r1[0] * (1.0f / DMODEL);
    float var = r2[0] * (1.0f / DMODEL) - mu * mu;
    float rstd = rsqrtf(var + LN_EPS);
    for (int i = tid; i < DMODEL; i += 256)
        y[base + i] = g[i] * (x[base + i] - mu) * rstd + b[i];
}

// ---------------- WMMA GEMM: C = act(A[M,K] @ B[K,N] (+bias)) (+res) ------
// 128x128 block tile, 256 threads = 8 wave32s (4 along M x 2 along N).
// Each wave owns a 32x64 patch = 2x4 v_wmma_f32_16x16x32_bf16 tiles.
// K stepped by 32 through bf16 LDS staging; next K-slab prefetched to L2.
#define LDS_STRIDE 40   // halves; 80B rows keep all fragment loads 16B aligned

template <bool HAS_BIAS, bool DO_GELU, bool RESIDUAL, bool B_TRANS>
__global__ __launch_bounds__(256)
void gemm_wmma_kernel(const float* __restrict__ A, const float* __restrict__ Bw,
                      const float* __restrict__ bias, const float* __restrict__ res,
                      float* __restrict__ C, int M, int N, int K) {
    __shared__ __align__(16) __bf16 sA[128 * LDS_STRIDE];   // [m][k]
    __shared__ __align__(16) __bf16 sB[128 * LDS_STRIDE];   // transposed: [n][k]

    const int tid  = threadIdx.x;
    const int lane = tid & 31;
    const int wave = tid >> 5;
    const int wm = (wave >> 1) * 32;        // wave row offset in tile (0..96)
    const int wn = (wave & 1) * 64;         // wave col offset in tile (0/64)
    const int l15 = lane & 15;
    const bool lo = lane < 16;
    const int m0 = blockIdx.y * 128;
    const int n0 = blockIdx.x * 128;

    v8f acc[2][4] = {};

    for (int k0 = 0; k0 < K; k0 += 32) {
        // stage A tile 128x32 (f32 -> bf16)
        for (int i = tid; i < 128 * 32; i += 256) {
            int m = i >> 5, kk = i & 31;
            sA[m * LDS_STRIDE + kk] = f2bf(A[(size_t)(m0 + m) * K + k0 + kk]);
        }
        // stage B tile 32x128, stored transposed as [n][k]
        for (int i = tid; i < 32 * 128; i += 256) {
            int kk = i >> 7, n = i & 127;
            int gn = n0 + n;
            float v = 0.f;
            if (gn < N)
                v = B_TRANS ? Bw[(size_t)gn * K + (k0 + kk)]
                            : Bw[(size_t)(k0 + kk) * N + gn];
            sB[n * LDS_STRIDE + kk] = f2bf(v);
        }
        // prefetch next K-slab into L2 while this one computes
        if (k0 + 32 < K) {
            if (tid < 128) {
                __builtin_prefetch(&A[(size_t)(m0 + tid) * K + k0 + 32], 0, 1);
            } else {
                int t = tid - 128;
                if (B_TRANS) {
                    int gn = n0 + t;
                    if (gn < N) __builtin_prefetch(&Bw[(size_t)gn * K + k0 + 32], 0, 1);
                } else {
                    int kk = t >> 2, c = (t & 3) * 32;
                    int gn = n0 + c;
                    if (gn < N)
                        __builtin_prefetch(&Bw[(size_t)(k0 + 32 + kk) * N + gn], 0, 1);
                }
            }
        }
        __syncthreads();

        // A fragments: lane row = l15; lanes<16: K 0-7 & 16-23, lanes>=16: 8-15 & 24-31
        bf16x16 afrag[2], bfrag[4];
        #pragma unroll
        for (int mi = 0; mi < 2; ++mi) {
            const __bf16* p = &sA[(wm + 16 * mi + l15) * LDS_STRIDE + (lo ? 0 : 8)];
            bf16x8 a_lo = *(const bf16x8*)p;
            bf16x8 a_hi = *(const bf16x8*)(p + 16);
            afrag[mi] = __builtin_shufflevector(a_lo, a_hi,
                0, 1, 2, 3, 4, 5, 6, 7, 8, 9, 10, 11, 12, 13, 14, 15);
        }
        // B fragments: lane col = l15; lanes<16: K 0-7 & 16-23, lanes>=16: 8-15 & 24-31
        #pragma unroll
        for (int ni = 0; ni < 4; ++ni) {
            const __bf16* p = &sB[(wn + 16 * ni + l15) * LDS_STRIDE + (lo ? 0 : 8)];
            bf16x8 b_lo = *(const bf16x8*)p;
            bf16x8 b_hi = *(const bf16x8*)(p + 16);
            bfrag[ni] = __builtin_shufflevector(b_lo, b_hi,
                0, 1, 2, 3, 4, 5, 6, 7, 8, 9, 10, 11, 12, 13, 14, 15);
        }
        #pragma unroll
        for (int mi = 0; mi < 2; ++mi)
            #pragma unroll
            for (int ni = 0; ni < 4; ++ni)
                acc[mi][ni] = __builtin_amdgcn_wmma_f32_16x16x32_bf16(
                    false, afrag[mi], false, bfrag[ni],
                    (short)0, acc[mi][ni], false, false);
        __syncthreads();
    }

    // C layout: VGPR r -> row r (lanes 0-15) or r+8 (lanes 16-31); col = l15
    const int rowadd = lo ? 0 : 8;
    #pragma unroll
    for (int mi = 0; mi < 2; ++mi) {
        #pragma unroll
        for (int ni = 0; ni < 4; ++ni) {
            int gn = n0 + wn + 16 * ni + l15;
            if (gn >= N) continue;
            #pragma unroll
            for (int r = 0; r < 8; ++r) {
                int gm = m0 + wm + 16 * mi + r + rowadd;
                float v = acc[mi][ni][r];
                if (HAS_BIAS) v += bias[gn];
                if (DO_GELU)  v = 0.5f * v * (1.0f + erff(v * 0.70710678118654752f));
                size_t oidx = (size_t)gm * N + gn;
                if (RESIDUAL) v += res[oidx];
                C[oidx] = v;
            }
        }
    }
}

// ---------------- causal attention, one wave32 per (b,h,q) ----------------
__global__ __launch_bounds__(128)
void attention_kernel(const float* __restrict__ q, const float* __restrict__ k,
                      const float* __restrict__ v, float* __restrict__ out) {
    __shared__ float sq[4][HDIM];
    const int tid = threadIdx.x;
    const int lane = tid & 31;
    const int w = tid >> 5;
    const int gw = blockIdx.x * 4 + w;          // 0 .. B*H*T-1
    const int b = gw / (NHEAD * SEQ);
    const int rem = gw % (NHEAD * SEQ);
    const int h = rem / SEQ;
    const int qi = rem % SEQ;

    const size_t qbase = (size_t)(b * SEQ + qi) * DMODEL + h * HDIM;
    sq[w][lane]      = q[qbase + lane];
    sq[w][lane + 32] = q[qbase + lane + 32];
    __syncthreads();

    const float scale = 0.125f;                 // 1/sqrt(64)
    float m = -INFINITY, l = 0.f, a0 = 0.f, a1 = 0.f;

    for (int kk = 0; kk <= qi; kk += 32) {
        int ki = kk + lane;
        float s = -INFINITY;
        if (ki <= qi) {
            const float* kr = k + (size_t)(b * SEQ + ki) * DMODEL + h * HDIM;
            float d = 0.f;
            #pragma unroll
            for (int i = 0; i < HDIM; ++i) d += sq[w][i] * kr[i];
            s = d * scale;
        }
        float cm = s;
        #pragma unroll
        for (int off = 16; off > 0; off >>= 1)
            cm = fmaxf(cm, __shfl_xor(cm, off, 32));
        float nm = fmaxf(m, cm);
        float corr = expf(m - nm);
        float p = (ki <= qi) ? expf(s - nm) : 0.f;
        float ps = p;
        #pragma unroll
        for (int off = 16; off > 0; off >>= 1)
            ps += __shfl_xor(ps, off, 32);
        l = l * corr + ps;
        a0 *= corr; a1 *= corr;
        for (int j = 0; j < 32; ++j) {
            int kj = kk + j;
            if (kj > qi) break;                 // uniform across the wave
            float pj = __shfl(p, j, 32);
            const float* vr = v + (size_t)(b * SEQ + kj) * DMODEL + h * HDIM;
            a0 += pj * vr[2 * lane];
            a1 += pj * vr[2 * lane + 1];
        }
        m = nm;
    }
    float inv = 1.f / l;
    size_t ob = (size_t)(b * SEQ + qi) * DMODEL + h * HDIM;
    out[ob + 2 * lane]     = a0 * inv;
    out[ob + 2 * lane + 1] = a1 * inv;
}

// ---------------- host driver ----------------
extern "C" void kernel_launch(void* const* d_in, const int* in_sizes, int n_in,
                              void* d_out, int out_size, void* d_ws, size_t ws_size,
                              hipStream_t stream) {
    const int*   in_idx = (const int*)  d_in[0];
    const float* wte    = (const float*)d_in[1];
    const float* wpe    = (const float*)d_in[2];
    const float* Wq     = (const float*)d_in[3];
    const float* Wk     = (const float*)d_in[4];
    const float* Wv     = (const float*)d_in[5];
    const float* Wo     = (const float*)d_in[6];
    const float* bo     = (const float*)d_in[7];
    const float* ln1_g  = (const float*)d_in[8];
    const float* ln1_b  = (const float*)d_in[9];
    const float* ln2_g  = (const float*)d_in[10];
    const float* ln2_b  = (const float*)d_in[11];
    const float* W1     = (const float*)d_in[12];
    const float* b1     = (const float*)d_in[13];
    const float* W2     = (const float*)d_in[14];
    const float* b2     = (const float*)d_in[15];
    const float* fg     = (const float*)d_in[16];
    const float* fb     = (const float*)d_in[17];
    float* logits = (float*)d_out;

    // workspace layout (f32)
    const size_t SZ = (size_t)ROWS * DMODEL;   // 1572864
    float* x   = (float*)d_ws;
    float* xn  = x   + SZ;
    float* qb  = xn  + SZ;
    float* kb  = qb  + SZ;
    float* vb  = kb  + SZ;
    float* ctx = vb  + SZ;
    float* hb  = ctx + SZ;                     // ROWS * FFDIM

    const dim3 gblk(256);
    const dim3 gD(DMODEL / 128, ROWS / 128);        // (6, 16)
    const dim3 gFF(FFDIM / 128, ROWS / 128);        // (24, 16)
    const dim3 gV((VOCAB + 127) / 128, ROWS / 128); // (393, 16)
    const int attnBlocks = (BATCH * NHEAD * SEQ) / 4; // 6144

    embed_kernel<<<ROWS, 256, 0, stream>>>(in_idx, wte, wpe, x);

    for (int i = 0; i < NLAYER; ++i) {
        const float* lg1 = ln1_g + (size_t)i * DMODEL;
        const float* lb1 = ln1_b + (size_t)i * DMODEL;
        const float* lg2 = ln2_g + (size_t)i * DMODEL;
        const float* lb2 = ln2_b + (size_t)i * DMODEL;
        const float* wq  = Wq + (size_t)i * DMODEL * DMODEL;
        const float* wk  = Wk + (size_t)i * DMODEL * DMODEL;
        const float* wv  = Wv + (size_t)i * DMODEL * DMODEL;
        const float* wo  = Wo + (size_t)i * DMODEL * DMODEL;
        const float* bob = bo + (size_t)i * DMODEL;
        const float* w1  = W1 + (size_t)i * DMODEL * FFDIM;
        const float* b1b = b1 + (size_t)i * FFDIM;
        const float* w2  = W2 + (size_t)i * FFDIM * DMODEL;
        const float* b2b = b2 + (size_t)i * DMODEL;

        layernorm_kernel<<<ROWS, 256, 0, stream>>>(x, lg1, lb1, xn);
        gemm_wmma_kernel<false, false, false, false><<<gD, gblk, 0, stream>>>(
            xn, wq, nullptr, nullptr, qb, ROWS, DMODEL, DMODEL);
        gemm_wmma_kernel<false, false, false, false><<<gD, gblk, 0, stream>>>(
            xn, wk, nullptr, nullptr, kb, ROWS, DMODEL, DMODEL);
        gemm_wmma_kernel<false, false, false, false><<<gD, gblk, 0, stream>>>(
            xn, wv, nullptr, nullptr, vb, ROWS, DMODEL, DMODEL);
        attention_kernel<<<attnBlocks, 128, 0, stream>>>(qb, kb, vb, ctx);
        gemm_wmma_kernel<true, false, true, false><<<gD, gblk, 0, stream>>>(
            ctx, wo, bob, x, x, ROWS, DMODEL, DMODEL);
        layernorm_kernel<<<ROWS, 256, 0, stream>>>(x, lg2, lb2, xn);
        gemm_wmma_kernel<true, true, false, false><<<gFF, gblk, 0, stream>>>(
            xn, w1, b1b, nullptr, hb, ROWS, FFDIM, DMODEL);
        gemm_wmma_kernel<true, false, true, false><<<gD, gblk, 0, stream>>>(
            hb, w2, b2b, x, x, ROWS, DMODEL, FFDIM);
    }

    layernorm_kernel<<<ROWS, 256, 0, stream>>>(x, fg, fb, xn);
    // logits = xn @ wte^T  (B transposed: B[k][n] = wte[n*D + k])
    gemm_wmma_kernel<false, false, false, true><<<gV, gblk, 0, stream>>>(
        xn, wte, nullptr, nullptr, logits, ROWS, VOCAB, DMODEL);
}
